// PFN_35459249995811
// MI455X (gfx1250) — matrix operand order repaired and checked by reading, not compiled, not run on hardware
//
#include <hip/hip_runtime.h>
#include <hip/hip_bf16.h>
#include <math.h>

#define SEQ    2048
#define BATCH  4
#define XDIM   16
#define DMODEL 256
#define NHEAD  8
#define DHEAD  32
#define DFF    512
#define NLAYER 2
#define MPOS   1024
#define ROWS   (SEQ * BATCH)   // 8192
#define NEGINF (-1000000000.0f)

typedef __attribute__((ext_vector_type(16))) _Float16 v16h;
typedef __attribute__((ext_vector_type(8)))  _Float16 v8h;
typedef __attribute__((ext_vector_type(4)))  _Float16 v4h;
typedef __attribute__((ext_vector_type(8)))  float    v8f;
typedef int v4i __attribute__((vector_size(16)));  // matches builtin param type

#if __has_builtin(__builtin_amdgcn_global_load_async_to_lds_b128) && \
    __has_builtin(__builtin_amdgcn_s_wait_asynccnt)
#define HAVE_ASYNC_LDS 1
#endif
#ifndef HAVE_ASYNC_LDS
#define HAVE_ASYNC_LDS 0
#endif

__device__ __forceinline__ v8f wmma_f16(v16h a, v16h b, v8f c) {
  // emits v_wmma_f32_16x16x32_f16
  return __builtin_amdgcn_wmma_f32_16x16x32_f16(false, a, false, b, (short)0, c, false, false);
}

// ---------------- cross-lane reductions: DPP16 butterflies (VALU, no LDS) --------------
#if __has_builtin(__builtin_amdgcn_update_dpp)
template <int CTRL>
__device__ __forceinline__ float dpp_mov_f32(float x) {
  return __builtin_bit_cast(
      float, __builtin_amdgcn_update_dpp(0, __builtin_bit_cast(int, x), CTRL, 0xF, 0xF, true));
}
__device__ __forceinline__ float row_max16(float v) {
  v = fmaxf(v, dpp_mov_f32<0xB1>(v));    // quad_perm [1,0,3,2]
  v = fmaxf(v, dpp_mov_f32<0x4E>(v));    // quad_perm [2,3,0,1]
  v = fmaxf(v, dpp_mov_f32<0x141>(v));   // row_half_mirror
  v = fmaxf(v, dpp_mov_f32<0x140>(v));   // row_mirror
  return v;
}
__device__ __forceinline__ float row_sum16(float v) {
  v += dpp_mov_f32<0xB1>(v);
  v += dpp_mov_f32<0x4E>(v);
  v += dpp_mov_f32<0x141>(v);
  v += dpp_mov_f32<0x140>(v);
  return v;
}
#else
__device__ __forceinline__ float row_max16(float v) {
#pragma unroll
  for (int m = 1; m < 16; m <<= 1) v = fmaxf(v, __shfl_xor(v, m, 32));
  return v;
}
__device__ __forceinline__ float row_sum16(float v) {
#pragma unroll
  for (int m = 1; m < 16; m <<= 1) v += __shfl_xor(v, m, 32);
  return v;
}
#endif
__device__ __forceinline__ float wave_sum32(float v) {
  v = row_sum16(v);
  v += __shfl_xor(v, 16, 32);
  return v;
}

// ---- A fragment: 16(M) x 32(K), source row-major ----
__device__ __forceinline__ v16h load_a_frag(const float* __restrict__ A, int lda, int row0, int k0) {
  int l = threadIdx.x & 31;
  const float* p = A + (size_t)(row0 + (l & 15)) * lda + k0 + ((l >> 4) << 4);
  v16h a;
#pragma unroll
  for (int i = 0; i < 8; ++i) a[i] = (_Float16)p[i];
#pragma unroll
  for (int i = 0; i < 8; ++i) a[8 + i] = (_Float16)p[8 + i];
  return a;
}

// ---- B fragment: 32(K) x 16(N) of W^T, W row-major (N x K) ----
__device__ __forceinline__ v16h load_bt_frag(const float* __restrict__ W, int ldw, int n0, int k0) {
  int l = threadIdx.x & 31;
  const float* p = W + (size_t)(n0 + (l & 15)) * ldw + k0 + ((l >> 4) << 4);
  v16h b;
#pragma unroll
  for (int i = 0; i < 16; ++i) b[i] = (_Float16)p[i];
  return b;
}

// ---- B fragment from an LDS f16 tile, 16 contiguous halfs per lane ----
__device__ __forceinline__ v16h load_lds_bt_frag(const _Float16* tile, int row, int kb16) {
  const v8h* p = (const v8h*)(tile + row * 32 + kb16);
  v8h lo = p[0];
  v8h hi = p[1];
  v16h r;
#pragma unroll
  for (int i = 0; i < 8; ++i) { r[i] = lo[i]; r[8 + i] = hi[i]; }
  return r;
}

// ==== GEMM: C = act(A @ W^T + bias); one wave per 32x32 tile; K templated, pipelined ====
template <int ACT, int K>  // ACT: 0 = none, 1 = exact GELU
__global__ void __launch_bounds__(256) gemm_kernel(const float* __restrict__ A,
                                                   const float* __restrict__ W,
                                                   const float* __restrict__ bias,
                                                   float* __restrict__ C,
                                                   int Mrows, int N) {
  int wave = threadIdx.x >> 5;
  int tile = blockIdx.x * (blockDim.x >> 5) + wave;
  int ntn  = N >> 5;
  int total = (Mrows >> 5) * ntn;
  if (tile >= total) return;
  int row0 = (tile / ntn) << 5;
  int n0   = (tile % ntn) << 5;
  int l = threadIdx.x & 31;

  v8f a00 = {}, a01 = {}, a10 = {}, a11 = {};
  v16h af0 = load_a_frag(A, K, row0, 0);
  v16h af1 = load_a_frag(A, K, row0 + 16, 0);
  v16h bf0 = load_bt_frag(W, K, n0, 0);
  v16h bf1 = load_bt_frag(W, K, n0 + 16, 0);
#pragma unroll
  for (int k = 0; k < K; k += 32) {
    v16h naf0, naf1, nbf0, nbf1;
    if (k + 32 < K) {
      if (k + 64 < K) {
        __builtin_prefetch(A + (size_t)(row0 + (l & 15)) * K + k + 64, 0, 1);
        __builtin_prefetch(W + (size_t)(n0 + (l & 15)) * K + k + 64, 0, 1);
      }
      naf0 = load_a_frag(A, K, row0, k + 32);
      naf1 = load_a_frag(A, K, row0 + 16, k + 32);
      nbf0 = load_bt_frag(W, K, n0, k + 32);
      nbf1 = load_bt_frag(W, K, n0 + 16, k + 32);
    }
    a00 = wmma_f16(af0, bf0, a00);
    a01 = wmma_f16(af0, bf1, a01);
    a10 = wmma_f16(af1, bf0, a10);
    a11 = wmma_f16(af1, bf1, a11);
    if (k + 32 < K) { af0 = naf0; af1 = naf1; bf0 = nbf0; bf1 = nbf1; }
  }
  int n  = l & 15;
  int mh = (l >> 4) << 3;
  float bv0 = bias[n0 + n];
  float bv1 = bias[n0 + 16 + n];
#pragma unroll
  for (int j = 0; j < 8; ++j) {
    float v0 = a00[j] + bv0, v1 = a01[j] + bv1;
    float v2 = a10[j] + bv0, v3 = a11[j] + bv1;
    if (ACT == 1) {
      v0 = 0.5f * v0 * (1.0f + erff(v0 * 0.70710678118654752f));
      v1 = 0.5f * v1 * (1.0f + erff(v1 * 0.70710678118654752f));
      v2 = 0.5f * v2 * (1.0f + erff(v2 * 0.70710678118654752f));
      v3 = 0.5f * v3 * (1.0f + erff(v3 * 0.70710678118654752f));
    }
    size_t r0 = (size_t)(row0 + mh + j) * N;
    size_t r1 = (size_t)(row0 + 16 + mh + j) * N;
    C[r0 + n0 + n]      = v0;
    C[r0 + n0 + 16 + n] = v1;
    C[r1 + n0 + n]      = v2;
    C[r1 + n0 + 16 + n] = v3;
  }
}

// ================= Flash attention, block-cooperative K/V staging ======================
// One block = one (b,h) x 8 query tiles (one per wave). Each 32-key tile is staged once
// per block into LDS f16 (K key-major, V transposed dh-major). With the gfx1250 async
// LDS-DMA available, raw f32 tiles are double-buffered via GLOBAL_LOAD_ASYNC_TO_LDS.
__global__ void __launch_bounds__(256) attn_kernel(const float* __restrict__ qkv,
                                                   float* __restrict__ o) {
#if HAVE_ASYNC_LDS
  __shared__ float kvscr[2][2048];       // [buf][K(1024) | V(1024)] f32 scratch, 16KB
#endif
  __shared__ _Float16 ktile[32 * 32];    // K f16 [key][dh], 2KB
  __shared__ _Float16 vtile[32 * 32];    // V f16 [dh][key] (transposed), 2KB
  __shared__ _Float16 psmem[8 * 16 * 32];// P staging, 1KB per wave

  int t    = threadIdx.x;
  int wave = t >> 5;
  int l    = t & 31;
  int bh = blockIdx.x >> 4;        // 0..31
  int qg = blockIdx.x & 15;        // query-tile group
  int b = bh >> 3, h = bh & 7;
  int qt = qg * 8 + wave;
  int s0 = qt << 4;
  _Float16* sm = psmem + wave * (16 * 32);

  int n    = l & 15;
  int mh   = (l >> 4) << 3;
  int kb16 = (l >> 4) << 4;
  int kb8  = (l >> 4) << 3;

  const int    ROWF = 3 * DMODEL;            // 768 floats per (seq,batch) row
  const size_t TOK  = (size_t)BATCH * ROWF;  // 3072 floats per sequence position

  // staging thread mapping (raw key-major copy): 4 floats per thread
  int skey = t >> 3;               // 0..31
  int sdh  = (t & 7) << 2;         // 0,4,...,28
  // V transpose mapping: 4 consecutive keys at one dh
  int vdh  = t >> 3;               // 0..31
  int vkey = (t & 7) << 2;         // 0,4,...,28
  const float* kgbase = qkv + (size_t)b * ROWF + DMODEL     + h * DHEAD + sdh + (size_t)skey * TOK;
  const float* vgbase = qkv + (size_t)b * ROWF + 2 * DMODEL + h * DHEAD;

  // Q fragment (A layout), K-dim = DH = 32
  v16h qf;
  {
    const float* p = qkv + (size_t)(s0 + n) * TOK + (size_t)b * ROWF + h * DHEAD + kb16;
#pragma unroll
    for (int i = 0; i < 8; ++i) qf[i] = (_Float16)p[i];
#pragma unroll
    for (int i = 0; i < 8; ++i) qf[8 + i] = (_Float16)p[8 + i];
  }

  v16h ones;
#pragma unroll
  for (int i = 0; i < 16; ++i) ones[i] = (_Float16)1.0f;

  float mrun[8];
  v8f O0 = {}, O1 = {}, Ls = {};
#pragma unroll
  for (int j = 0; j < 8; ++j) mrun[j] = -1e30f;

  const float scale = 0.17677669529663688f;  // 1/sqrt(DH)

#if HAVE_ASYNC_LDS
  // prologue: async-load tile 0 raw f32 into buffer 0
  __builtin_amdgcn_global_load_async_to_lds_b128(
      (v4i*)(kgbase), (v4i*)(&kvscr[0][skey * 32 + sdh]), 0, 0);
  __builtin_amdgcn_global_load_async_to_lds_b128(
      (v4i*)(vgbase + sdh + (size_t)skey * TOK), (v4i*)(&kvscr[0][1024 + skey * 32 + sdh]), 0, 0);
#endif

  for (int it = 0; it < 32; ++it) {  // keys [0, 1024)
#if HAVE_ASYNC_LDS
    if (it + 1 < 32) {  // double buffer: issue next tile, then wait for current
      int nb = (it + 1) & 1;
      const float* kg = kgbase + (size_t)((it + 1) * 32) * TOK;
      const float* vg = vgbase + sdh + (size_t)(skey + (it + 1) * 32) * TOK;
      __builtin_amdgcn_global_load_async_to_lds_b128(
          (v4i*)(kg), (v4i*)(&kvscr[nb][skey * 32 + sdh]), 0, 0);
      __builtin_amdgcn_global_load_async_to_lds_b128(
          (v4i*)(vg), (v4i*)(&kvscr[nb][1024 + skey * 32 + sdh]), 0, 0);
      __builtin_amdgcn_s_wait_asynccnt(2);
    } else {
      __builtin_amdgcn_s_wait_asynccnt(0);
    }
#endif
    __syncthreads();  // scratch for tile `it` ready; previous consume finished

    // cooperative convert: f32 -> f16 tiles (K key-major, V transposed)
    {
#if HAVE_ASYNC_LDS
      const float* ksrc = &kvscr[it & 1][skey * 32 + sdh];
      const float* vsrc = &kvscr[it & 1][1024];
      v4h kpk;
#pragma unroll
      for (int i = 0; i < 4; ++i) kpk[i] = (_Float16)ksrc[i];
      *(v4h*)(ktile + skey * 32 + sdh) = kpk;
      v4h vpk;
#pragma unroll
      for (int i = 0; i < 4; ++i) vpk[i] = (_Float16)vsrc[(vkey + i) * 32 + vdh];
      *(v4h*)(vtile + vdh * 32 + vkey) = vpk;
#else
      const float* ksrc = kgbase + (size_t)(it * 32) * TOK;
      v4h kpk;
#pragma unroll
      for (int i = 0; i < 4; ++i) kpk[i] = (_Float16)ksrc[i];
      *(v4h*)(ktile + skey * 32 + sdh) = kpk;
      v4h vpk;
#pragma unroll
      for (int i = 0; i < 4; ++i)
        vpk[i] = (_Float16)vgbase[vdh + (size_t)(it * 32 + vkey + i) * TOK];
      *(v4h*)(vtile + vdh * 32 + vkey) = vpk;
#endif
    }
    __syncthreads();  // f16 tiles ready

    // fragments from LDS (contiguous ds_load_b128 pairs)
    v16h kf0 = load_lds_bt_frag(ktile, n, kb16);        // keys n,      dh kb16..+15
    v16h kf1 = load_lds_bt_frag(ktile, 16 + n, kb16);   // keys 16+n
    v16h vf0 = load_lds_bt_frag(vtile, n, kb16);        // dh n,        keys kb16..+15
    v16h vf1 = load_lds_bt_frag(vtile, 16 + n, kb16);   // dh 16+n

    v8f z = {};
    v8f S0 = wmma_f16(qf, kf0, z);
    v8f S1 = wmma_f16(qf, kf1, z);

#pragma unroll
    for (int j = 0; j < 8; ++j) {
      float s0v = S0[j] * scale;
      float s1v = S1[j] * scale;
      float tmax = row_max16(fmaxf(s0v, s1v));
      float mnew = fmaxf(mrun[j], tmax);
      float c = __expf(mrun[j] - mnew);
      mrun[j] = mnew;
      float p0 = __expf(s0v - mnew);
      float p1 = __expf(s1v - mnew);
      O0[j] *= c; O1[j] *= c; Ls[j] *= c;
      sm[(mh + j) * 32 + n]      = (_Float16)p0;
      sm[(mh + j) * 32 + 16 + n] = (_Float16)p1;
    }
    __builtin_amdgcn_fence(__ATOMIC_ACQ_REL, "wavefront");

    v16h pf;
    {
      const v8h* p = (const v8h*)(sm + (l & 15) * 32 + kb16);
      v8h lo = p[0];
      v8h hi = p[1];
#pragma unroll
      for (int i = 0; i < 8; ++i) { pf[i] = lo[i]; pf[8 + i] = hi[i]; }
    }
    __builtin_amdgcn_fence(__ATOMIC_ACQ_REL, "wavefront");

    O0 = wmma_f16(pf, vf0, O0);
    O1 = wmma_f16(pf, vf1, O1);
    Ls = wmma_f16(pf, ones, Ls);
  }

  // diagonal self-key tile (block-uniform predicate; per-wave direct global loads)
  if (s0 >= MPOS) {
    v16h kf0;
    {
      const float* p = qkv + (size_t)(s0 + n) * TOK + (size_t)b * ROWF + DMODEL + h * DHEAD + kb16;
#pragma unroll
      for (int i = 0; i < 16; ++i) kf0[i] = (_Float16)p[i];
    }
    v8f z = {};
    v8f S0 = wmma_f16(qf, kf0, z);
#pragma unroll
    for (int j = 0; j < 8; ++j) {
      float s0v = S0[j] * scale + ((n == mh + j) ? 0.0f : NEGINF);
      float tmax = row_max16(s0v);
      float mnew = fmaxf(mrun[j], tmax);
      float c = __expf(mrun[j] - mnew);
      mrun[j] = mnew;
      float p0 = __expf(s0v - mnew);
      O0[j] *= c; O1[j] *= c; Ls[j] *= c;
      sm[(mh + j) * 32 + n]      = (_Float16)p0;
      sm[(mh + j) * 32 + 16 + n] = (_Float16)0.0f;
    }
    __builtin_amdgcn_fence(__ATOMIC_ACQ_REL, "wavefront");
    v16h pf;
    {
      const v8h* p = (const v8h*)(sm + (l & 15) * 32 + kb16);
      v8h lo = p[0];
      v8h hi = p[1];
#pragma unroll
      for (int i = 0; i < 8; ++i) { pf[i] = lo[i]; pf[8 + i] = hi[i]; }
    }
    __builtin_amdgcn_fence(__ATOMIC_ACQ_REL, "wavefront");
    v16h vf0, vf1;
#pragma unroll
    for (int i = 0; i < 16; ++i) {
      int key = s0 + kb16 + i;
      if (key > SEQ - 1) key = SEQ - 1;  // clamped rows multiply zero P entries
      const float* pv = qkv + (size_t)key * TOK + (size_t)b * ROWF + 2 * DMODEL + h * DHEAD + n;
      vf0[i] = (_Float16)pv[0];
      vf1[i] = (_Float16)pv[16];
    }
    O0 = wmma_f16(pf, vf0, O0);
    O1 = wmma_f16(pf, vf1, O1);
    Ls = wmma_f16(pf, ones, Ls);
  }

  // normalize + store
#pragma unroll
  for (int j = 0; j < 8; ++j) {
    float inv = 1.0f / Ls[j];
    float* po = o + (size_t)(s0 + mh + j) * (BATCH * DMODEL) + b * DMODEL + h * DHEAD;
    po[n]      = O0[j] * inv;
    po[16 + n] = O1[j] * inv;
  }
}

// ================= encoder ====================
__global__ void encoder_kernel(const float* __restrict__ x, const float* __restrict__ y,
                               const float* __restrict__ xw, const float* __restrict__ xb,
                               const float* __restrict__ yw, const float* __restrict__ yb,
                               float* __restrict__ tokens) {
  int idx = blockIdx.x * blockDim.x + threadIdx.x;
  if (idx >= ROWS * DMODEL) return;
  int d = idx & (DMODEL - 1);
  int r = idx >> 8;
  int s = r >> 2;
  float acc = xb[d];
  const float* xr = x + (size_t)r * XDIM;
  const float* wr = xw + (size_t)d * XDIM;
#pragma unroll
  for (int k = 0; k < XDIM; ++k) acc += xr[k] * wr[k];
  if (s < MPOS) acc += y[r] * yw[d] + yb[d];
  tokens[idx] = acc;
}

// ================= residual add + LayerNorm (one wave per row) ================
__global__ void add_ln_kernel(const float* __restrict__ X, const float* __restrict__ R,
                              const float* __restrict__ g, const float* __restrict__ bb,
                              float* __restrict__ Out) {
  int row = blockIdx.x * (blockDim.x >> 5) + (threadIdx.x >> 5);
  if (row >= ROWS) return;
  int lane = threadIdx.x & 31;
  float v[8];
  float s = 0.f, s2 = 0.f;
#pragma unroll
  for (int i = 0; i < 8; ++i) {
    int c = lane + (i << 5);
    float t = X[(size_t)row * DMODEL + c] + R[(size_t)row * DMODEL + c];
    v[i] = t; s += t; s2 += t * t;
  }
  s  = wave_sum32(s);
  s2 = wave_sum32(s2);
  float mu  = s * (1.0f / DMODEL);
  float var = s2 * (1.0f / DMODEL) - mu * mu;
  float inv = rsqrtf(var + 1e-5f);
#pragma unroll
  for (int i = 0; i < 8; ++i) {
    int c = lane + (i << 5);
    Out[(size_t)row * DMODEL + c] = (v[i] - mu) * inv * g[c] + bb[c];
  }
}

// ================= final head projection to scalar (one wave per row) ==================
__global__ void head2_kernel(const float* __restrict__ hid, const float* __restrict__ w2,
                             const float* __restrict__ b2, float* __restrict__ out, int rows) {
  int row = blockIdx.x * (blockDim.x >> 5) + (threadIdx.x >> 5);
  if (row >= rows) return;
  int lane = threadIdx.x & 31;
  float s = 0.f;
  for (int i = lane; i < DFF; i += 32) s += hid[(size_t)row * DFF + i] * w2[i];
  s = wave_sum32(s);
  if (lane == 0) out[row] = s + b2[0];
}

extern "C" void kernel_launch(void* const* d_in, const int* in_sizes, int n_in,
                              void* d_out, int out_size, void* d_ws, size_t ws_size,
                              hipStream_t stream) {
  (void)in_sizes; (void)n_in; (void)out_size; (void)ws_size;
  const float* x         = (const float*)d_in[0];
  const float* y         = (const float*)d_in[1];
  // d_in[2] = single_eval_pos (int scalar) == MPOS, fixed by problem setup
  const float* xenc_w    = (const float*)d_in[3];
  const float* xenc_b    = (const float*)d_in[4];
  const float* yenc_w    = (const float*)d_in[5];
  const float* yenc_b    = (const float*)d_in[6];
  const float* in_proj_w = (const float*)d_in[7];
  const float* in_proj_b = (const float*)d_in[8];
  const float* out_proj_w= (const float*)d_in[9];
  const float* out_proj_b= (const float*)d_in[10];
  const float* ln1_g     = (const float*)d_in[11];
  const float* ln1_b     = (const float*)d_in[12];
  const float* lin1_w    = (const float*)d_in[13];
  const float* lin1_b    = (const float*)d_in[14];
  const float* lin2_w    = (const float*)d_in[15];
  const float* lin2_b    = (const float*)d_in[16];
  const float* ln2_g     = (const float*)d_in[17];
  const float* ln2_b     = (const float*)d_in[18];
  const float* head_w1   = (const float*)d_in[19];
  const float* head_b1   = (const float*)d_in[20];
  const float* head_w2   = (const float*)d_in[21];
  const float* head_b2   = (const float*)d_in[22];

  float* ws  = (float*)d_ws;
  float* h   = ws;                               // 8192*256
  float* big = ws + (size_t)ROWS * DMODEL;       // 8192*768 (qkv / proj / ff / hid)
  float* o   = big + (size_t)ROWS * 3 * DMODEL;  // 8192*256

  encoder_kernel<<<(ROWS * DMODEL + 255) / 256, 256, 0, stream>>>(
      x, y, xenc_w, xenc_b, yenc_w, yenc_b, h);

  for (int l = 0; l < NLAYER; ++l) {
    {  // qkv = h @ Wqkv^T + b : (8192 x 768), K=256
      int tiles = (ROWS / 32) * ((3 * DMODEL) / 32);
      gemm_kernel<0, DMODEL><<<(tiles + 7) / 8, 256, 0, stream>>>(
          h, in_proj_w + (size_t)l * 3 * DMODEL * DMODEL,
          in_proj_b + (size_t)l * 3 * DMODEL, big, ROWS, 3 * DMODEL);
    }
    attn_kernel<<<512, 256, 0, stream>>>(big, o);
    {  // out proj -> big, K=256
      int tiles = (ROWS / 32) * (DMODEL / 32);
      gemm_kernel<0, DMODEL><<<(tiles + 7) / 8, 256, 0, stream>>>(
          o, out_proj_w + (size_t)l * DMODEL * DMODEL,
          out_proj_b + (size_t)l * DMODEL, big, ROWS, DMODEL);
    }
    add_ln_kernel<<<(ROWS + 7) / 8, 256, 0, stream>>>(
        big, h, ln1_g + (size_t)l * DMODEL, ln1_b + (size_t)l * DMODEL, h);
    {  // ff1 (gelu) -> big, K=256
      int tiles = (ROWS / 32) * (DFF / 32);
      gemm_kernel<1, DMODEL><<<(tiles + 7) / 8, 256, 0, stream>>>(
          h, lin1_w + (size_t)l * DFF * DMODEL, lin1_b + (size_t)l * DFF,
          big, ROWS, DFF);
    }
    {  // ff2 -> o, K=512
      int tiles = (ROWS / 32) * (DMODEL / 32);
      gemm_kernel<0, DFF><<<(tiles + 7) / 8, 256, 0, stream>>>(
          big, lin2_w + (size_t)l * DMODEL * DFF, lin2_b + (size_t)l * DMODEL,
          o, ROWS, DMODEL);
    }
    add_ln_kernel<<<(ROWS + 7) / 8, 256, 0, stream>>>(
        o, h, ln2_g + (size_t)l * DMODEL, ln2_b + (size_t)l * DMODEL, h);
  }

  // head: rows with s >= M (4096 rows), K=256
  const int qrows = (SEQ - MPOS) * BATCH;
  float* hid = big;
  {
    int tiles = (qrows / 32) * (DFF / 32);
    gemm_kernel<1, DMODEL><<<(tiles + 7) / 8, 256, 0, stream>>>(
        h + (size_t)MPOS * BATCH * DMODEL, head_w1, head_b1, hid, qrows, DFF);
  }
  head2_kernel<<<(qrows + 7) / 8, 256, 0, stream>>>(hid, head_w2, head_b2,
                                                   (float*)d_out, qrows);
}
